// Detr3DDecoderLayer_68650757259995
// MI455X (gfx1250) — matrix operand order repaired and checked by reading, not compiled, not run on hardware
//
#include <hip/hip_runtime.h>
#include <hip/hip_bf16.h>

#define B_  2
#define Q_  900
#define C_  256
#define NH_ 8
#define NC_ 6
#define NL_ 3
#define DH_ 32
#define QP_ 928          // Q padded to multiple of 32 for attention key chunks
#define MQ_ (B_*Q_)      // 1800
#define MP_ 1920         // MQ_ padded to multiple of 128 (GEMM M-block)

typedef __attribute__((ext_vector_type(16))) __bf16 v16bf;
typedef __attribute__((ext_vector_type(8)))  float  v8f;

union Frag { v16bf v; unsigned u[8]; };

__device__ __forceinline__ unsigned short f2bf(float f) {
    union { float f; unsigned u; } c; c.f = f;
    unsigned r = (c.u + 0x7FFFu + ((c.u >> 16) & 1u)) >> 16;   // round-nearest-even
    return (unsigned short)r;
}

// ASYNCcnt-tracked global -> LDS transfers (CDNA5 async copy path)
__device__ __forceinline__ void async_load_b128(unsigned lds_off, const void* gptr) {
    asm volatile("global_load_async_to_lds_b128 %0, %1, off"
                 :: "v"(lds_off), "v"(gptr) : "memory");
}
__device__ __forceinline__ void async_load_b32(unsigned lds_off, const void* gptr) {
    asm volatile("global_load_async_to_lds_b32 %0, %1, off"
                 :: "v"(lds_off), "v"(gptr) : "memory");
}
__device__ __forceinline__ void wait_async() {
    asm volatile("s_wait_asynccnt 0x0" ::: "memory");
}
// Generic flat pointer to a __shared__ object -> 32-bit LDS byte offset.
__device__ __forceinline__ unsigned lds_off32(const void* p) {
    return (unsigned)(uintptr_t)p;
}

// ---------------------------------------------------------------------------
// fp32 -> bf16 bulk convert
// ---------------------------------------------------------------------------
__global__ void convert_bf16_kernel(const float* __restrict__ src,
                                    unsigned short* __restrict__ dst, int n) {
    int i = blockIdx.x * 256 + threadIdx.x;
    if (i < n) dst[i] = f2bf(src[i]);
}

// q = query + query_pos (bf16), plus bf16 copy of query (V input of MHA)
__global__ void add_pos_kernel(const float* __restrict__ query,
                               const float* __restrict__ qpos,
                               unsigned short* __restrict__ q_bf,
                               unsigned short* __restrict__ query_bf, int n) {
    int i = blockIdx.x * 256 + threadIdx.x;
    if (i < n) {
        float qq = query[i];
        q_bf[i]     = f2bf(qq + qpos[i]);
        query_bf[i] = f2bf(qq);
    }
}

// ---------------------------------------------------------------------------
// Generic tiled WMMA GEMM:  D[M,N] = act(A[Mp,K] @ W[N,K]^T + bias)
// A rows padded to a multiple of 128 (garbage rows allowed; outputs guarded).
// A tile: 512x b128 async transfers; W tile: 256x b32 async transfers (no EXEC
// masking in the hot loop). Block = 256 threads = 8 waves; each wave computes
// one 16x16 tile of a 128x16 block with v_wmma_f32_16x16x32_bf16.
// ---------------------------------------------------------------------------
__global__ __launch_bounds__(256)
void gemm_bf16_kernel(const unsigned short* __restrict__ A,
                      const unsigned short* __restrict__ W,
                      const float* __restrict__ bias,
                      float* __restrict__ D,
                      unsigned short* __restrict__ Dbf,
                      int M, int N, int K, int ldD, int act) {
    __shared__ unsigned At[128 * 16];  // 128 rows x 32 k (16 bf16-pairs/row)
    __shared__ unsigned Wt[16 * 16];   // 16 n x 32 k
    const int n0   = blockIdx.x * 16;
    const int m0   = blockIdx.y * 128;
    const int tid  = threadIdx.x;
    const int wave = tid >> 5, lane = tid & 31;
    const int half = lane >> 4, lm = lane & 15;
    const int Ku   = K >> 1;
    const unsigned* A_u = (const unsigned*)A;
    const unsigned* W_u = (const unsigned*)W;

    // per-thread transfer slots (constant across K loop)
    const int a_row0 = (tid + 0)   >> 2, a_c40 = (tid + 0)   & 3;
    const int a_row1 = (tid + 256) >> 2, a_c41 = (tid + 256) & 3;
    const unsigned aoff0 = lds_off32(&At[(tid + 0)   * 4]);
    const unsigned aoff1 = lds_off32(&At[(tid + 256) * 4]);
    const int w_row = tid >> 4, w_c = tid & 15;          // one dword per thread
    const unsigned woff = lds_off32(&Wt[tid]);

    v8f acc = {};
    for (int kk = 0; kk < K; kk += 32) {
        const int base_u = kk >> 1;
        // A tile: 2 x 16B per thread, no bounds check (M padded)
        async_load_b128(aoff0, A_u + (size_t)(m0 + a_row0) * Ku + base_u + a_c40 * 4);
        async_load_b128(aoff1, A_u + (size_t)(m0 + a_row1) * Ku + base_u + a_c41 * 4);
        // W tile: 4B per thread, all lanes active
        async_load_b32(woff, W_u + (size_t)(n0 + w_row) * Ku + base_u + w_c);
        wait_async();
        __syncthreads();

        Frag a, b;
        const int am = wave * 16 + lm;
        #pragma unroll
        for (int i = 0; i < 8; ++i)
            a.u[i] = At[am * 16 + half * 8 + i];            // K = half*16 + 2i
        #pragma unroll
        for (int i = 0; i < 8; ++i)
            b.u[i] = Wt[lm * 16 + half * 8 + i];            // K = half*16 + 2i
        acc = __builtin_amdgcn_wmma_f32_16x16x32_bf16(false, a.v, false, b.v,
                                                      (short)0, acc, false, false);
        __syncthreads();
    }
    const int col = n0 + lm;
    const float bi = bias ? bias[col] : 0.f;               // hoisted out of loop
    #pragma unroll
    for (int r = 0; r < 8; ++r) {
        int row = m0 + wave * 16 + half * 8 + r;
        if (row < M) {
            float v = acc[r] + bi;
            if (act) v = fmaxf(v, 0.f);
            D[(size_t)row * ldD + col] = v;
            if (Dbf) Dbf[(size_t)row * ldD + col] = f2bf(v);
        }
    }
}

// ---------------------------------------------------------------------------
// Split qkv fp32 [MQ,768]: qh,kh -> [B*NH, QP, 32] bf16 (zero padded rows),
//                          vh    -> [B*NH, 32, QP] bf16 TRANSPOSED (K-pairs contiguous)
// ---------------------------------------------------------------------------
__global__ void repack_qkv_kernel(const float* __restrict__ qkv,
                                  unsigned short* __restrict__ qh,
                                  unsigned short* __restrict__ kh,
                                  unsigned short* __restrict__ vh) {
    int idx = blockIdx.x * 256 + threadIdx.x;
    const int TOT = B_ * NH_ * QP_ * DH_;
    if (idx >= TOT) return;
    int d  = idx & 31;
    int qp = (idx >> 5) % QP_;
    int bh = idx / (QP_ * DH_);
    int b = bh >> 3, h = bh & 7;
    unsigned short qv = 0, kv = 0, vv = 0;
    if (qp < Q_) {
        size_t rb = (size_t)(b * Q_ + qp) * (3 * C_);
        qv = f2bf(qkv[rb +          h * 32 + d]);
        kv = f2bf(qkv[rb + C_     + h * 32 + d]);
        vv = f2bf(qkv[rb + 2 * C_ + h * 32 + d]);
    }
    qh[idx] = qv; kh[idx] = kv;
    vh[(size_t)bh * DH_ * QP_ + (size_t)d * QP_ + qp] = vv;   // transposed
}

// ---------------------------------------------------------------------------
// Flash attention: one wave = one (b,h, 16-query tile). 8 waves / block.
// scores: 2x v_wmma per 32-key chunk (K=dh=32). P transposed via per-wave LDS.
// ---------------------------------------------------------------------------
__global__ __launch_bounds__(256)
void flash_attn_kernel(const unsigned short* __restrict__ qh,
                       const unsigned short* __restrict__ kh,
                       const unsigned short* __restrict__ vh,
                       float* __restrict__ o) {   // [B,Q,C], col = h*32+d
    __shared__ unsigned short Plds[8][16 * 32];
    const int wave = threadIdx.x >> 5, lane = threadIdx.x & 31;
    const int half = lane >> 4, lm = lane & 15;
    const int tile = blockIdx.x * 8 + wave;        // 912 tiles total, grid=114
    const int TPB  = 57;                           // ceil(900/16)
    const int bh = tile / TPB, qt = tile % TPB;
    const int q0 = qt * 16;

    Frag aq;
    const unsigned* qbase = (const unsigned*)qh + ((size_t)bh * QP_ + q0) * 16;
    #pragma unroll
    for (int i = 0; i < 8; ++i)
        aq.u[i] = qbase[lm * 16 + half * 8 + i];
    const unsigned* kb  = (const unsigned*)kh + (size_t)bh * QP_ * 16;
    const unsigned* vtu = (const unsigned*)vh + (size_t)bh * DH_ * (QP_ / 2);

    float mrow[8], lrow[8];
    #pragma unroll
    for (int r = 0; r < 8; ++r) { mrow[r] = -1e30f; lrow[r] = 0.f; }
    v8f o0 = {}, o1 = {};
    const float scale = 0.17677669529663687f;      // 1/sqrt(32)

    for (int k0 = 0; k0 < Q_; k0 += 32) {
        Frag bk0, bk1;
        #pragma unroll
        for (int i = 0; i < 8; ++i) {
            bk0.u[i] = kb[(size_t)(k0 + lm)      * 16 + half * 8 + i];
            bk1.u[i] = kb[(size_t)(k0 + 16 + lm) * 16 + half * 8 + i];
        }
        v8f s0 = {}, s1 = {};
        s0 = __builtin_amdgcn_wmma_f32_16x16x32_bf16(false, aq.v, false, bk0.v,
                                                     (short)0, s0, false, false);
        s1 = __builtin_amdgcn_wmma_f32_16x16x32_bf16(false, aq.v, false, bk1.v,
                                                     (short)0, s1, false, false);
        const int c0 = k0 + lm, c1 = k0 + 16 + lm;
        float p0[8], p1[8];
        #pragma unroll
        for (int r = 0; r < 8; ++r) {
            float v0 = (c0 < Q_) ? s0[r] * scale : -1e30f;
            float v1 = (c1 < Q_) ? s1[r] * scale : -1e30f;
            float mx = fmaxf(v0, v1);
            #pragma unroll
            for (int off = 1; off < 16; off <<= 1)
                mx = fmaxf(mx, __shfl_xor(mx, off, 16));   // row max within half
            float mnew = fmaxf(mrow[r], mx);
            float a  = __expf(mrow[r] - mnew);
            float e0 = __expf(v0 - mnew), e1 = __expf(v1 - mnew);
            float sm = e0 + e1;
            #pragma unroll
            for (int off = 1; off < 16; off <<= 1)
                sm += __shfl_xor(sm, off, 16);
            lrow[r] = lrow[r] * a + sm;
            mrow[r] = mnew;
            p0[r] = e0; p1[r] = e1;
            o0[r] *= a; o1[r] *= a;
        }
        // C-layout -> A-layout transpose through per-wave LDS
        unsigned short* P = Plds[wave];
        #pragma unroll
        for (int r = 0; r < 8; ++r) {
            int row = half * 8 + r;
            P[row * 32 + lm]      = f2bf(p0[r]);
            P[row * 32 + 16 + lm] = f2bf(p1[r]);
        }
        asm volatile("s_wait_dscnt 0x0" ::: "memory");     // same-wave DS RAW fence
        Frag ap;
        const unsigned* Pu = (const unsigned*)P;
        #pragma unroll
        for (int i = 0; i < 8; ++i)
            ap.u[i] = Pu[lm * 16 + half * 8 + i];
        // V^T layout: element (k,n) at vt[n*QP + k]; K-pairs are one dword
        Frag bv0, bv1;
        const int kq = k0 >> 1;
        #pragma unroll
        for (int i = 0; i < 8; ++i) {
            bv0.u[i] = vtu[(size_t)lm        * (QP_ / 2) + kq + half * 8 + i];
            bv1.u[i] = vtu[(size_t)(16 + lm) * (QP_ / 2) + kq + half * 8 + i];
        }
        o0 = __builtin_amdgcn_wmma_f32_16x16x32_bf16(false, ap.v, false, bv0.v,
                                                     (short)0, o0, false, false);
        o1 = __builtin_amdgcn_wmma_f32_16x16x32_bf16(false, ap.v, false, bv1.v,
                                                     (short)0, o1, false, false);
    }
    const int b = bh / NH_, h = bh % NH_;
    #pragma unroll
    for (int r = 0; r < 8; ++r) {
        int q = q0 + half * 8 + r;
        if (q < Q_) {
            float inv = 1.f / lrow[r];
            size_t base = ((size_t)b * Q_ + q) * C_ + h * 32;
            o[base + lm]      = o0[r] * inv;
            o[base + 16 + lm] = o1[r] * inv;
        }
    }
}

// ---------------------------------------------------------------------------
// LayerNorm over C=256: out = act((in0+in1+in2 - mu)*rsqrt(var+eps)*g + b)
// ---------------------------------------------------------------------------
__global__ __launch_bounds__(256)
void ln_kernel(const float* __restrict__ in0, const float* __restrict__ in1,
               const float* __restrict__ in2, const float* __restrict__ g,
               const float* __restrict__ bt, float* __restrict__ out,
               unsigned short* __restrict__ outbf, int relu) {
    int row = blockIdx.x, c = threadIdx.x;
    size_t idx = (size_t)row * C_ + c;
    float v = in0[idx];
    if (in1) v += in1[idx];
    if (in2) v += in2[idx];
    __shared__ float s1[256], s2[256];
    s1[c] = v; s2[c] = v * v;
    __syncthreads();
    for (int off = 128; off > 0; off >>= 1) {
        if (c < off) { s1[c] += s1[c + off]; s2[c] += s2[c + off]; }
        __syncthreads();
    }
    float mu  = s1[0] * (1.f / 256.f);
    float var = s2[0] * (1.f / 256.f) - mu * mu;
    float y = (v - mu) * rsqrtf(var + 1e-5f) * g[c] + bt[c];
    if (relu) y = fmaxf(y, 0.f);
    out[idx] = y;
    if (outbf) outbf[idx] = f2bf(y);
}

// aw = sigmoid(x1 @ attW^T + attb), 18 cols — tiny, plain VALU
__global__ __launch_bounds__(256)
void aw_kernel(const float* __restrict__ x1, const float* __restrict__ attW,
               const float* __restrict__ attb, float* __restrict__ aw) {
    int bq = blockIdx.x, tid = threadIdx.x;
    __shared__ float xr[256];
    xr[tid] = x1[(size_t)bq * C_ + tid];
    __syncthreads();
    if (tid < NC_ * NL_) {
        float s = attb[tid];
        const float* w = attW + tid * C_;
        for (int c = 0; c < C_; ++c) s += xr[c] * w[c];
        aw[(size_t)bq * (NC_ * NL_) + tid] = 1.f / (1.f + __expf(-s));
    }
}

// pe1raw = inv_sigmoid(ref) @ peW1^T + peb1   (K=3, plain VALU)
__global__ __launch_bounds__(256)
void pe1_kernel(const float* __restrict__ ref, const float* __restrict__ W,
                const float* __restrict__ bias, float* __restrict__ out) {
    int bq = blockIdx.x, c = threadIdx.x;
    __shared__ float iv[3];
    if (c < 3) {
        float r = ref[(size_t)bq * 3 + c];
        float a  = fminf(fmaxf(r, 1e-5f), 1.f);
        float b2 = fminf(fmaxf(1.f - r, 1e-5f), 1.f);
        iv[c] = __logf(a / b2);
    }
    __syncthreads();
    out[(size_t)bq * C_ + c] =
        bias[c] + iv[0] * W[c * 3] + iv[1] * W[c * 3 + 1] + iv[2] * W[c * 3 + 2];
}

// ---------------------------------------------------------------------------
// Camera projection + bilinear sample + level/camera fusion, one block per (b,q)
// ---------------------------------------------------------------------------
__global__ __launch_bounds__(256)
void sample_fuse_kernel(const float* __restrict__ ref, const float* __restrict__ l2i,
                        const float* __restrict__ f0, const float* __restrict__ f1,
                        const float* __restrict__ f2, const float* __restrict__ aw,
                        float* __restrict__ fused, unsigned short* __restrict__ fusedbf) {
    const int HH[3] = {116, 58, 29}, WW[3] = {200, 100, 50};
    int bq = blockIdx.x, b = bq / Q_, q = bq % Q_, tid = threadIdx.x;
    __shared__ float gx[NC_], gy[NC_]; __shared__ int msk[NC_];
    __shared__ const float* basep[18];
    __shared__ int o00[18], o01[18], o10[18], o11[18], hwv[18];
    __shared__ float w00[18], w01[18], w10[18], w11[18];

    if (tid < NC_) {
        int n = tid;
        float rx = ref[(size_t)bq * 3], ry = ref[(size_t)bq * 3 + 1], rz = ref[(size_t)bq * 3 + 2];
        float x3 = rx * 102.4f - 51.2f;
        float y3 = ry * 102.4f - 51.2f;
        float z3 = rz * 8.0f - 5.0f;
        const float* L = l2i + (size_t)(b * NC_ + n) * 16;
        float c0 = L[0] * x3 + L[1] * y3 + L[2]  * z3 + L[3];
        float c1 = L[4] * x3 + L[5] * y3 + L[6]  * z3 + L[7];
        float c2 = L[8] * x3 + L[9] * y3 + L[10] * z3 + L[11];
        int m = c2 > 1e-5f;
        float dz = fmaxf(c2, 1e-5f);
        float gxx = (c0 / dz) / 1600.f * 2.f - 1.f;
        float gyy = (c1 / dz) / 928.f  * 2.f - 1.f;
        m = m && (gxx > -1.f) && (gxx < 1.f) && (gyy > -1.f) && (gyy < 1.f);
        gx[n] = gxx; gy[n] = gyy; msk[n] = m;
    }
    __syncthreads();
    if (tid < 18) {
        int n = tid / 3, l = tid % 3;
        int H = HH[l], W = WW[l];
        float x = (gx[n] + 1.f) * 0.5f * W - 0.5f;
        float y = (gy[n] + 1.f) * 0.5f * H - 0.5f;
        float xf = floorf(x), yf = floorf(y);
        int x0 = (int)xf, y0 = (int)yf;
        float wx = x - xf, wy = y - yf;
        float awm = msk[n] ? aw[(size_t)bq * 18 + n * 3 + l] : 0.f;
        auto cl = [](int v, int hi) { return v < 0 ? 0 : (v > hi ? hi : v); };
        int vx0 = (x0 >= 0) && (x0 < W),     vx1 = (x0 + 1 >= 0) && (x0 + 1 < W);
        int vy0 = (y0 >= 0) && (y0 < H),     vy1 = (y0 + 1 >= 0) && (y0 + 1 < H);
        int cx0 = cl(x0, W - 1), cx1 = cl(x0 + 1, W - 1);
        int cy0 = cl(y0, H - 1), cy1 = cl(y0 + 1, H - 1);
        o00[tid] = cy0 * W + cx0; o01[tid] = cy0 * W + cx1;
        o10[tid] = cy1 * W + cx0; o11[tid] = cy1 * W + cx1;
        w00[tid] = awm * (1.f - wx) * (1.f - wy) * (vx0 && vy0);
        w01[tid] = awm * wx * (1.f - wy) * (vx1 && vy0);
        w10[tid] = awm * (1.f - wx) * wy * (vx0 && vy1);
        w11[tid] = awm * wx * wy * (vx1 && vy1);
        const float* f = (l == 0) ? f0 : ((l == 1) ? f1 : f2);
        basep[tid] = f + (size_t)(b * NC_ + n) * C_ * (H * W);
        hwv[tid] = H * W;
    }
    __syncthreads();
    int c = tid;
    float acc = 0.f;
    for (int t = 0; t < 18; ++t) {
        const float* bp = basep[t] + (size_t)c * hwv[t];
        acc += w00[t] * bp[o00[t]] + w01[t] * bp[o01[t]]
             + w10[t] * bp[o10[t]] + w11[t] * bp[o11[t]];
    }
    fused[(size_t)bq * C_ + c] = acc;
    fusedbf[(size_t)bq * C_ + c] = f2bf(acc);
}

// ---------------------------------------------------------------------------
extern "C" void kernel_launch(void* const* d_in, const int* in_sizes, int n_in,
                              void* d_out, int out_size, void* d_ws, size_t ws_size,
                              hipStream_t stream) {
    const float* query = (const float*)d_in[0];
    const float* qpos  = (const float*)d_in[1];
    const float* ref   = (const float*)d_in[2];
    const float* l2i   = (const float*)d_in[3];
    const float* f0    = (const float*)d_in[4];
    const float* f1    = (const float*)d_in[5];
    const float* f2    = (const float*)d_in[6];
    const float* Wqkv  = (const float*)d_in[7];
    const float* bqkv  = (const float*)d_in[8];
    const float* Wo    = (const float*)d_in[9];
    const float* bo    = (const float*)d_in[10];
    const float* attW  = (const float*)d_in[11];
    const float* attb  = (const float*)d_in[12];
    const float* outW  = (const float*)d_in[13];
    const float* outb  = (const float*)d_in[14];
    const float* peW1  = (const float*)d_in[15];
    const float* peb1  = (const float*)d_in[16];
    const float* peg1  = (const float*)d_in[17];
    const float* pebt1 = (const float*)d_in[18];
    const float* peW2  = (const float*)d_in[19];
    const float* peb2  = (const float*)d_in[20];
    const float* peg2  = (const float*)d_in[21];
    const float* pebt2 = (const float*)d_in[22];
    const float* ffW1  = (const float*)d_in[23];
    const float* ffb1  = (const float*)d_in[24];
    const float* ffW2  = (const float*)d_in[25];
    const float* ffb2  = (const float*)d_in[26];
    const float* n1g = (const float*)d_in[27];
    const float* n1b = (const float*)d_in[28];
    const float* n2g = (const float*)d_in[29];
    const float* n2b = (const float*)d_in[30];
    const float* n3g = (const float*)d_in[31];
    const float* n3b = (const float*)d_in[32];
    float* out = (float*)d_out;

    char* p = (char*)d_ws;
    auto alloc = [&](size_t bytes) -> void* {
        void* r = (void*)p; p += (bytes + 255) & ~(size_t)255; return r;
    };
    // A-side bf16 buffers padded to MP_ rows so GEMM staging needs no guards
    unsigned short* q_bf     = (unsigned short*)alloc((size_t)MP_ * C_ * 2);
    unsigned short* query_bf = (unsigned short*)alloc((size_t)MP_ * C_ * 2);
    unsigned short* Wqkv_bf  = (unsigned short*)alloc(3 * C_ * C_ * 2);
    unsigned short* Wo_bf    = (unsigned short*)alloc(C_ * C_ * 2);
    unsigned short* outW_bf  = (unsigned short*)alloc(C_ * C_ * 2);
    unsigned short* peW2_bf  = (unsigned short*)alloc(C_ * C_ * 2);
    unsigned short* ffW1_bf  = (unsigned short*)alloc(4 * C_ * C_ * 2);
    unsigned short* ffW2_bf  = (unsigned short*)alloc(4 * C_ * C_ * 2);
    float*          qkv      = (float*)alloc((size_t)MQ_ * 3 * C_ * 4);
    unsigned short* qh       = (unsigned short*)alloc((size_t)B_*NH_*QP_*DH_ * 2);
    unsigned short* kh       = (unsigned short*)alloc((size_t)B_*NH_*QP_*DH_ * 2);
    unsigned short* vh       = (unsigned short*)alloc((size_t)B_*NH_*QP_*DH_ * 2);
    float*          o_f32    = (float*)alloc((size_t)MQ_ * C_ * 4);
    unsigned short* o_bf     = (unsigned short*)alloc((size_t)MP_ * C_ * 2);
    float*          sa       = (float*)alloc((size_t)MQ_ * C_ * 4);
    float*          x1       = (float*)alloc((size_t)MQ_ * C_ * 4);
    unsigned short* x1_bf    = (unsigned short*)alloc((size_t)MP_ * C_ * 2);
    float*          aw       = (float*)alloc((size_t)MQ_ * 18 * 4);
    float*          fused    = (float*)alloc((size_t)MQ_ * C_ * 4);
    unsigned short* fused_bf = (unsigned short*)alloc((size_t)MP_ * C_ * 2);
    float*          outraw   = (float*)alloc((size_t)MQ_ * C_ * 4);
    float*          pe1raw   = (float*)alloc((size_t)MQ_ * C_ * 4);
    float*          pe1      = (float*)alloc((size_t)MQ_ * C_ * 4);
    unsigned short* pe1_bf   = (unsigned short*)alloc((size_t)MP_ * C_ * 2);
    float*          pe2raw   = (float*)alloc((size_t)MQ_ * C_ * 4);
    float*          pe       = (float*)alloc((size_t)MQ_ * C_ * 4);
    float*          x2       = (float*)alloc((size_t)MQ_ * C_ * 4);
    unsigned short* x2_bf    = (unsigned short*)alloc((size_t)MP_ * C_ * 2);
    float*          ff1      = (float*)alloc((size_t)MQ_ * 4 * C_ * 4);
    unsigned short* ff1_bf   = (unsigned short*)alloc((size_t)MP_ * 4 * C_ * 2);
    float*          ff2      = (float*)alloc((size_t)MQ_ * C_ * 4);

    auto cvt = [&](const float* s, unsigned short* d, int n) {
        convert_bf16_kernel<<<(n + 255) / 256, 256, 0, stream>>>(s, d, n);
    };
    cvt(Wqkv, Wqkv_bf, 3 * C_ * C_);
    cvt(Wo,   Wo_bf,   C_ * C_);
    cvt(outW, outW_bf, C_ * C_);
    cvt(peW2, peW2_bf, C_ * C_);
    cvt(ffW1, ffW1_bf, 4 * C_ * C_);
    cvt(ffW2, ffW2_bf, 4 * C_ * C_);

    const int NE = MQ_ * C_;
    add_pos_kernel<<<(NE + 255) / 256, 256, 0, stream>>>(query, qpos, q_bf, query_bf, NE);

    dim3 blk(256);
    const int gy = MP_ / 128;   // 15
    // QKV projection: Q,K from (query+pos), V from query
    gemm_bf16_kernel<<<dim3(512 / 16, gy), blk, 0, stream>>>(
        q_bf, Wqkv_bf, bqkv, qkv, nullptr, MQ_, 512, C_, 3 * C_, 0);
    gemm_bf16_kernel<<<dim3(256 / 16, gy), blk, 0, stream>>>(
        query_bf, Wqkv_bf + (size_t)2 * C_ * C_, bqkv + 2 * C_,
        qkv + 2 * C_, nullptr, MQ_, 256, C_, 3 * C_, 0);

    const int TOT = B_ * NH_ * QP_ * DH_;
    repack_qkv_kernel<<<(TOT + 255) / 256, 256, 0, stream>>>(qkv, qh, kh, vh);

    flash_attn_kernel<<<114, 256, 0, stream>>>(qh, kh, vh, o_f32);
    cvt(o_f32, o_bf, NE);

    gemm_bf16_kernel<<<dim3(16, gy), blk, 0, stream>>>(
        o_bf, Wo_bf, bo, sa, nullptr, MQ_, C_, C_, C_, 0);
    ln_kernel<<<MQ_, 256, 0, stream>>>(query, sa, nullptr, n1g, n1b, x1, x1_bf, 0);

    aw_kernel<<<MQ_, 256, 0, stream>>>(x1, attW, attb, aw);
    sample_fuse_kernel<<<MQ_, 256, 0, stream>>>(ref, l2i, f0, f1, f2, aw, fused, fused_bf);

    gemm_bf16_kernel<<<dim3(16, gy), blk, 0, stream>>>(
        fused_bf, outW_bf, outb, outraw, nullptr, MQ_, C_, C_, C_, 0);

    pe1_kernel<<<MQ_, 256, 0, stream>>>(ref, peW1, peb1, pe1raw);
    ln_kernel<<<MQ_, 256, 0, stream>>>(pe1raw, nullptr, nullptr, peg1, pebt1, pe1, pe1_bf, 1);
    gemm_bf16_kernel<<<dim3(16, gy), blk, 0, stream>>>(
        pe1_bf, peW2_bf, peb2, pe2raw, nullptr, MQ_, C_, C_, C_, 0);
    ln_kernel<<<MQ_, 256, 0, stream>>>(pe2raw, nullptr, nullptr, peg2, pebt2, pe, nullptr, 1);

    // x2 = LN(x1 + outraw + pe)
    ln_kernel<<<MQ_, 256, 0, stream>>>(x1, outraw, pe, n2g, n2b, x2, x2_bf, 0);

    // FFN
    gemm_bf16_kernel<<<dim3(1024 / 16, gy), blk, 0, stream>>>(
        x2_bf, ffW1_bf, ffb1, ff1, ff1_bf, MQ_, 4 * C_, C_, 4 * C_, 1);
    gemm_bf16_kernel<<<dim3(16, gy), blk, 0, stream>>>(
        ff1_bf, ffW2_bf, ffb2, ff2, nullptr, MQ_, C_, 4 * C_, C_, 0);
    ln_kernel<<<MQ_, 256, 0, stream>>>(x2, ff2, nullptr, n3g, n3b, out, nullptr, 0);
}
